// ACSeg_36653250904770
// MI455X (gfx1250) — compile-verified, matrix operand order checked by reading, not run on hardware
//
#include <hip/hip_runtime.h>

#define Bn   8
#define Tn   784
#define Dn   384
#define Hn   6
#define DHn  64
#define Kn   5
#define DFFn 1536
#define Llay 3
#define KC   128   // k-chunk staged in LDS (3 chunks cover Dn=384)

typedef __attribute__((ext_vector_type(16))) __bf16 bf16x16;
typedef __attribute__((ext_vector_type(8)))  __bf16 bf16x8;
typedef __attribute__((ext_vector_type(8)))  float  v8f;
typedef int v4i_vs __attribute__((vector_size(16)));   // matches async-builtin param

#define WMMA_BF16(a, b, c) \
  __builtin_amdgcn_wmma_f32_16x16x32_bf16(false, (a), false, (b), (short)0, (c), false, false)

__device__ inline unsigned short f2bf(float f) {
  unsigned int u = __float_as_uint(f);
  unsigned int r = (u + 0x7FFFu + ((u >> 16) & 1u)) >> 16;
  return (unsigned short)r;
}

// ---- gfx1250 async global->LDS staging (ASYNCcnt) with portable fallback ----
#if __has_builtin(__builtin_amdgcn_global_load_async_to_lds_b128)
#define ACSEG_HAVE_ASYNC 1
#endif

__device__ inline void cp16(unsigned short* lds, const unsigned short* g) {
#if defined(ACSEG_HAVE_ASYNC)
  __builtin_amdgcn_global_load_async_to_lds_b128(
      (__attribute__((address_space(1))) v4i_vs*)g,
      (__attribute__((address_space(3))) v4i_vs*)lds,
      0, 0);
#else
  *reinterpret_cast<bf16x8*>(lds) = *reinterpret_cast<const bf16x8*>(g);
#endif
}

__device__ inline void wait_async() {
#if defined(ACSEG_HAVE_ASYNC)
#if __has_builtin(__builtin_amdgcn_s_wait_asynccnt)
  __builtin_amdgcn_s_wait_asynccnt(0);
#else
  asm volatile("s_wait_asynccnt 0x0" ::: "memory");
#endif
#endif
}

// Read a 16x32 bf16 fragment from an LDS slab (row-major, leading dim KC).
// A-fragment lane layout per CDNA5 ISA 7.12.2; for X @ X^T / X @ W^T both
// operands use the identical pattern.
__device__ inline bf16x16 lds_frag(const unsigned short* sl, int row0, int k0) {
  int lane = threadIdx.x & 31;
  const unsigned short* p = sl + (row0 + (lane & 15)) * KC + k0 + ((lane & 16) ? 8 : 0);
  bf16x8 lo = *reinterpret_cast<const bf16x8*>(p);       // K .. K+7
  bf16x8 hi = *reinterpret_cast<const bf16x8*>(p + 16);  // K+16 .. K+23
  bf16x16 f;
#pragma unroll
  for (int i = 0; i < 8; ++i) { f[i] = lo[i]; f[i + 8] = hi[i]; }
  return f;
}

__device__ inline void store_tile(float* Y, int ldY, int rbase, int col,
                                  const v8f& a, float bv) {
#pragma unroll
  for (int r = 0; r < 8; ++r) Y[(size_t)(rbase + r) * ldY + col] = a[r] + bv;
}

// ---------------- prep: bf16 copies of x, x/||x||, relu(x)/||relu(x)|| ------
__global__ void acseg_prep(const float* __restrict__ x,
                           unsigned short* __restrict__ xb,
                           unsigned short* __restrict__ xn,
                           unsigned short* __restrict__ xcn) {
  int row = blockIdx.x;
  int tid = threadIdx.x;
  const float* xr = x + (size_t)row * Dn;
  __shared__ float r1[128], r2[128];
  float s = 0.f, sc = 0.f;
  for (int d = tid; d < Dn; d += 128) {
    float v = xr[d]; s += v * v;
    float c = fmaxf(v, 0.f); sc += c * c;
  }
  r1[tid] = s; r2[tid] = sc; __syncthreads();
  for (int st = 64; st > 0; st >>= 1) {
    if (tid < st) { r1[tid] += r1[tid + st]; r2[tid] += r2[tid + st]; }
    __syncthreads();
  }
  float rn = 1.f / fmaxf(sqrtf(r1[0]), 1e-8f);
  float rc = 1.f / fmaxf(sqrtf(r2[0]), 1e-8f);
  for (int d = tid; d < Dn; d += 128) {
    float v = xr[d]; float c = fmaxf(v, 0.f);
    size_t o = (size_t)row * Dn + d;
    xb[o] = f2bf(v); xn[o] = f2bf(v * rn); xcn[o] = f2bf(c * rc);
  }
}

// ---------------- weight transpose + f32->bf16 ------------------------------
__global__ void acseg_convT(const float* __restrict__ w, unsigned short* __restrict__ wT,
                            int Kd, int N) {
  int idx = blockIdx.x * 256 + threadIdx.x;
  if (idx >= Kd * N) return;
  int k = idx / N, n = idx % N;
  wT[(size_t)n * Kd + k] = f2bf(w[idx]);
}

// ---------------- WMMA GEMM: Y = Xbf @ W + bias, 64x64 tile per block -------
// A: M x Dn bf16; BT: N x Dn bf16 (transposed weights). grid=(N/64, M/64),
// 128 threads (4 waves). LDS-staged k-chunks; wave w does rows i0+16w,
// reusing its A-fragment across 4 B-fragments (4 WMMAs per 5 frag reads).
__global__ __launch_bounds__(128) void acseg_gemm_wmma(
    const unsigned short* __restrict__ A, const unsigned short* __restrict__ BT,
    const float* __restrict__ bias, float* __restrict__ Y) {
  __shared__ alignas(16) unsigned short Asl[64 * KC];
  __shared__ alignas(16) unsigned short Bsl[64 * KC];
  int i0 = blockIdx.y * 64, j0 = blockIdx.x * 64;
  int tid = threadIdx.x, lane = tid & 31, w = tid >> 5;
  v8f acc[4] = {};
  for (int kc = 0; kc < Dn; kc += KC) {
    __syncthreads();
    for (int t = tid; t < (64 * KC) / 8; t += 128) {
      int r = t >> 4, c = (t & 15) << 3;
      cp16(&Asl[r * KC + c], A  + (size_t)(i0 + r) * Dn + kc + c);
      cp16(&Bsl[r * KC + c], BT + (size_t)(j0 + r) * Dn + kc + c);
    }
    wait_async();
    __syncthreads();
#pragma unroll
    for (int k = 0; k < KC; k += 32) {
      bf16x16 af = lds_frag(Asl, 16 * w, k);
#pragma unroll
      for (int jt = 0; jt < 4; ++jt) {
        bf16x16 bf = lds_frag(Bsl, 16 * jt, k);
        acc[jt] = WMMA_BF16(af, bf, acc[jt]);
      }
    }
  }
  int rbase = i0 + 16 * w + ((lane & 16) ? 8 : 0);
#pragma unroll
  for (int jt = 0; jt < 4; ++jt) {
    int col = j0 + 16 * jt + (lane & 15);
    store_tile(Y, Dn, rbase, col, acc[jt], bias[col]);
  }
}

// ---------------- affinity A = x̂ x̂^T (batch0 raw, else relu) ---------------
// Same 64x64 cooperative tiling; T=784 = 12*64+16 handled with uniform
// 16-tile guards so EXEC stays all-ones around every WMMA.
__global__ __launch_bounds__(128) void acseg_affinity(
    const unsigned short* __restrict__ xn, const unsigned short* __restrict__ xcn,
    float* __restrict__ W) {
  __shared__ alignas(16) unsigned short Asl[64 * KC];
  __shared__ alignas(16) unsigned short Bsl[64 * KC];
  int b = blockIdx.z;
  int i0 = blockIdx.y * 64, j0 = blockIdx.x * 64;
  const unsigned short* base = (b == 0 ? xn : xcn) + (size_t)b * Tn * Dn;
  int tid = threadIdx.x, lane = tid & 31, w = tid >> 5;
  bool rowOK = (i0 + 16 * w) < Tn;
  v8f acc[4] = {};
  for (int kc = 0; kc < Dn; kc += KC) {
    __syncthreads();
    for (int t = tid; t < (64 * KC) / 8; t += 128) {
      int r = t >> 4, c = (t & 15) << 3;
      int ra = i0 + r; ra = ra < Tn ? ra : Tn - 1;   // clamp edge rows
      int rb = j0 + r; rb = rb < Tn ? rb : Tn - 1;
      cp16(&Asl[r * KC + c], base + (size_t)ra * Dn + kc + c);
      cp16(&Bsl[r * KC + c], base + (size_t)rb * Dn + kc + c);
    }
    wait_async();
    __syncthreads();
    if (rowOK) {
#pragma unroll
      for (int k = 0; k < KC; k += 32) {
        bf16x16 af = lds_frag(Asl, 16 * w, k);
#pragma unroll
        for (int jt = 0; jt < 4; ++jt) {
          if (j0 + 16 * jt < Tn) {
            bf16x16 bf = lds_frag(Bsl, 16 * jt, k);
            acc[jt] = WMMA_BF16(af, bf, acc[jt]);
          }
        }
      }
    }
  }
  if (rowOK) {
    float* Wb = W + (size_t)b * Tn * Tn;
    int rbase = i0 + 16 * w + ((lane & 16) ? 8 : 0);
#pragma unroll
    for (int jt = 0; jt < 4; ++jt) {
      if (j0 + 16 * jt < Tn) {
        int col = j0 + 16 * jt + (lane & 15);
        store_tile(Wb, Tn, rbase, col, acc[jt], 0.f);
      }
    }
  }
}

// ---------------- deterministic row sums -> Kdeg ----------------------------
__global__ void acseg_rowsum(const float* __restrict__ W, float* __restrict__ Kdeg) {
  int i = blockIdx.x, b = blockIdx.y, tid = threadIdx.x;
  __shared__ float red[256];
  const float* row = W + ((size_t)b * Tn + i) * Tn;
  float s = 0.f;
  for (int j = tid; j < Tn; j += 256) s += row[j];
  red[tid] = s; __syncthreads();
  for (int st = 128; st > 0; st >>= 1) {
    if (tid < st) red[tid] += red[tid + st];
    __syncthreads();
  }
  if (tid == 0) Kdeg[b * Tn + i] = red[0];
}

__global__ void acseg_finalize(const float* __restrict__ Kdeg,
                               float* __restrict__ m_out, float* __restrict__ scale) {
  int b = blockIdx.x, tid = threadIdx.x;
  __shared__ float r1[256], r2[256];
  float s = 0.f, s2 = 0.f;
  for (int t = tid; t < Tn; t += 256) {
    float v = Kdeg[b * Tn + t]; s += v; s2 += v * v;
  }
  r1[tid] = s; r2[tid] = s2; __syncthreads();
  for (int st = 128; st > 0; st >>= 1) {
    if (tid < st) { r1[tid] += r1[tid + st]; r2[tid] += r2[tid + st]; }
    __syncthreads();
  }
  if (tid == 0) { m_out[b] = r1[0]; scale[b] = r2[0] / r1[0]; }
}

__global__ void acseg_wscale(float* __restrict__ W, const float* __restrict__ scale) {
  size_t idx = (size_t)blockIdx.x * 256 + threadIdx.x;
  if (idx >= (size_t)Bn * Tn * Tn) return;
  int b = (int)(idx / ((size_t)Tn * Tn));
  W[idx] *= scale[b];
}

// ---------------- small linear: Y = X@W + bias (+add0)(+add1) (relu?) -------
__global__ void acseg_lin(const float* __restrict__ X, const float* __restrict__ Wt,
                          const float* __restrict__ bias, const float* __restrict__ add0,
                          const float* __restrict__ add1, float* __restrict__ Y,
                          int M, int Kd, int N, int relu) {
  int idx = blockIdx.x * 256 + threadIdx.x;
  if (idx >= M * N) return;
  int m = idx / N, n = idx % N;
  float a = bias ? bias[n] : 0.f;
  const float* xr = X + (size_t)m * Kd;
  for (int k = 0; k < Kd; ++k) a += xr[k] * Wt[(size_t)k * N + n];
  if (add0) a += add0[idx];
  if (add1) a += add1[idx];
  if (relu) a = fmaxf(a, 0.f);
  Y[idx] = a;
}

// ---------------- layernorm over last dim (rows of length Dn) ---------------
__global__ void acseg_ln(const float* __restrict__ xin, const float* __restrict__ g,
                         const float* __restrict__ bta, float* __restrict__ y) {
  int row = blockIdx.x, tid = threadIdx.x;
  const float* xr = xin + (size_t)row * Dn;
  __shared__ float red[128];
  float s = 0.f;
  for (int d = tid; d < Dn; d += 128) s += xr[d];
  red[tid] = s; __syncthreads();
  for (int st = 64; st > 0; st >>= 1) { if (tid < st) red[tid] += red[tid + st]; __syncthreads(); }
  float mu = red[0] / Dn; __syncthreads();
  float v = 0.f;
  for (int d = tid; d < Dn; d += 128) { float t = xr[d] - mu; v += t * t; }
  red[tid] = v; __syncthreads();
  for (int st = 64; st > 0; st >>= 1) { if (tid < st) red[tid] += red[tid + st]; __syncthreads(); }
  float rstd = rsqrtf(red[0] / Dn + 1e-5f);
  for (int d = tid; d < Dn; d += 128)
    y[(size_t)row * Dn + d] = (xr[d] - mu) * rstd * g[d] + bta[d];
}

// ---------------- attention: 5 queries x Tk keys, per (b,h,qi) block --------
__global__ void acseg_attn(const float* __restrict__ q, const float* __restrict__ k,
                           const float* __restrict__ v, float* __restrict__ o, int Tk) {
  int qi = blockIdx.x, h = blockIdx.y, b = blockIdx.z, tid = threadIdx.x;
  __shared__ float s[Tn];
  __shared__ float red[256];
  const float* qrow = q + ((size_t)(b * Kn + qi)) * Dn + h * DHn;
  const float* kb = k + (size_t)b * Tk * Dn + h * DHn;
  const float* vb = v + (size_t)b * Tk * Dn + h * DHn;
  float scale = rsqrtf((float)Dn);      // reference scales by sqrt(d_model)
  float lmax = -1e30f;
  for (int t = tid; t < Tk; t += 256) {
    float acc = 0.f;
    for (int d = 0; d < DHn; ++d) acc += qrow[d] * kb[(size_t)t * Dn + d];
    acc *= scale; s[t] = acc; lmax = fmaxf(lmax, acc);
  }
  red[tid] = lmax; __syncthreads();
  for (int st = 128; st > 0; st >>= 1) { if (tid < st) red[tid] = fmaxf(red[tid], red[tid + st]); __syncthreads(); }
  float gmax = red[0]; __syncthreads();
  float lsum = 0.f;
  for (int t = tid; t < Tk; t += 256) { float e = __expf(s[t] - gmax); s[t] = e; lsum += e; }
  red[tid] = lsum; __syncthreads();
  for (int st = 128; st > 0; st >>= 1) { if (tid < st) red[tid] += red[tid + st]; __syncthreads(); }
  float denom = red[0]; __syncthreads();
  int d = tid & 63, g = tid >> 6;
  float acc = 0.f;
  for (int t = g; t < Tk; t += 4) acc += s[t] * vb[(size_t)t * Dn + d];
  red[tid] = acc; __syncthreads();
  if (g == 0) {
    float r = red[d] + red[64 + d] + red[128 + d] + red[192 + d];
    o[((size_t)(b * Kn + qi)) * Dn + h * DHn + d] = r / denom;
  }
}

// ---------------- misc small kernels ----------------------------------------
__global__ void acseg_initout(const float* __restrict__ proto, float* __restrict__ outbuf) {
  int i = blockIdx.x * 256 + threadIdx.x;
  if (i >= Kn * Dn) return;
  float v = proto[i];
  for (int b = 0; b < Bn; ++b) outbuf[(size_t)b * Kn * Dn + i] = v;
}

__global__ void acseg_normrows(const float* __restrict__ in, float* __restrict__ out) {
  int row = blockIdx.x, tid = threadIdx.x;
  const float* xr = in + (size_t)row * Dn;
  __shared__ float red[128];
  float s = 0.f;
  for (int d = tid; d < Dn; d += 128) { float v = xr[d]; s += v * v; }
  red[tid] = s; __syncthreads();
  for (int st = 64; st > 0; st >>= 1) { if (tid < st) red[tid] += red[tid + st]; __syncthreads(); }
  float rn = 1.f / fmaxf(sqrtf(red[0]), 1e-8f);
  for (int d = tid; d < Dn; d += 128) out[(size_t)row * Dn + d] = xr[d] * rn;
}

__global__ void acseg_sassign(const float* __restrict__ x, const float* __restrict__ concn,
                              float* __restrict__ S, float* __restrict__ assign_out) {
  int t = blockIdx.x, b = blockIdx.y, tid = threadIdx.x;
  const float* xr = x + ((size_t)b * Tn + t) * Dn;
  __shared__ float red[128];
  __shared__ float dots[Kn];
  float s = 0.f;
  for (int d = tid; d < Dn; d += 128) { float v = xr[d]; s += v * v; }
  red[tid] = s; __syncthreads();
  for (int st = 64; st > 0; st >>= 1) { if (tid < st) red[tid] += red[tid + st]; __syncthreads(); }
  float rn = 1.f / fmaxf(sqrtf(red[0]), 1e-8f);
  __syncthreads();
  for (int kk = 0; kk < Kn; ++kk) {
    const float* cr = concn + ((size_t)b * Kn + kk) * Dn;
    float a = 0.f;
    for (int d = tid; d < Dn; d += 128) a += xr[d] * cr[d];
    red[tid] = a; __syncthreads();
    for (int st = 64; st > 0; st >>= 1) { if (tid < st) red[tid] += red[tid + st]; __syncthreads(); }
    if (tid == 0) dots[kk] = fmaxf(red[0] * rn, 0.f);
    __syncthreads();
  }
  if (tid == 0) {
    int best = 0; float bv = dots[0];
    for (int kk = 1; kk < Kn; ++kk) if (dots[kk] > bv) { bv = dots[kk]; best = kk; }
    for (int kk = 0; kk < Kn; ++kk) S[((size_t)b * Tn + t) * Kn + kk] = dots[kk];
    assign_out[(size_t)b * Tn + t] = (float)best;
  }
}

__global__ void acseg_delta(const float* __restrict__ S, float* __restrict__ delta) {
  int i = blockIdx.x, b = blockIdx.y, tid = threadIdx.x;
  __shared__ float si[Kn];
  if (tid < Kn) si[tid] = S[((size_t)b * Tn + i) * Kn + tid];
  __syncthreads();
  for (int j = tid; j < Tn; j += 256) {
    const float* sj = S + ((size_t)b * Tn + j) * Kn;
    float mx = si[0] * sj[0];
#pragma unroll
    for (int kk = 1; kk < Kn; ++kk) mx = fmaxf(mx, si[kk] * sj[kk]);
    delta[((size_t)b * Tn + i) * Tn + j] = mx;
  }
}

// ============================================================================
extern "C" void kernel_launch(void* const* d_in, const int* in_sizes, int n_in,
                              void* d_out, int out_size, void* d_ws, size_t ws_size,
                              hipStream_t stream) {
  const float* x     = (const float*)d_in[0];
  const float* proto = (const float*)d_in[1];
  auto L_in = [&](int l, int j) -> const float* { return (const float*)d_in[2 + l * 26 + j]; };

  char* wsp = (char*)d_ws;
  size_t off = 0;
  auto alloc = [&](size_t bytes) -> void* {
    void* p = wsp + off; off += (bytes + 255) & ~(size_t)255; return p;
  };

  unsigned short* xb  = (unsigned short*)alloc((size_t)Bn * Tn * Dn * 2);
  unsigned short* xn  = (unsigned short*)alloc((size_t)Bn * Tn * Dn * 2);
  unsigned short* xcn = (unsigned short*)alloc((size_t)Bn * Tn * Dn * 2);
  unsigned short* wTk = (unsigned short*)alloc((size_t)Dn * Dn * 2);
  unsigned short* wTv = (unsigned short*)alloc((size_t)Dn * Dn * 2);
  float* kbuf  = (float*)alloc((size_t)Bn * Tn * Dn * 4);
  float* vbuf  = (float*)alloc((size_t)Bn * Tn * Dn * 4);
  float* outbuf= (float*)alloc((size_t)Bn * Kn * Dn * 4);
  float* qbuf  = (float*)alloc((size_t)Bn * Kn * Dn * 4);
  float* obuf  = (float*)alloc((size_t)Bn * Kn * Dn * 4);
  float* t1    = (float*)alloc((size_t)Bn * Kn * Dn * 4);
  float* x1    = (float*)alloc((size_t)Bn * Kn * Dn * 4);
  float* q2    = (float*)alloc((size_t)Bn * Kn * Dn * 4);
  float* k2    = (float*)alloc((size_t)Bn * Kn * Dn * 4);
  float* v2    = (float*)alloc((size_t)Bn * Kn * Dn * 4);
  float* o2    = (float*)alloc((size_t)Bn * Kn * Dn * 4);
  float* t2    = (float*)alloc((size_t)Bn * Kn * Dn * 4);
  float* x2    = (float*)alloc((size_t)Bn * Kn * Dn * 4);
  float* hb    = (float*)alloc((size_t)Bn * Kn * DFFn * 4);
  float* t3    = (float*)alloc((size_t)Bn * Kn * Dn * 4);
  float* concn = (float*)alloc((size_t)Bn * Kn * Dn * 4);
  float* Kdeg  = (float*)alloc((size_t)Bn * Tn * 4);
  float* scl   = (float*)alloc((size_t)Bn * 4);
  float* S     = (float*)alloc((size_t)Bn * Tn * Kn * 4);

  float* Wout = (float*)d_out;
  float* Dout = Wout + (size_t)Bn * Tn * Tn;
  float* Aout = Dout + (size_t)Bn * Tn * Tn;
  float* Mout = Aout + (size_t)Bn * Tn;

  acseg_prep<<<Bn * Tn, 128, 0, stream>>>(x, xb, xn, xcn);
  acseg_initout<<<(Kn * Dn + 255) / 256, 256, 0, stream>>>(proto, outbuf);

  int MN  = Bn * Kn * Dn;
  int MN2 = Bn * Kn * DFFn;
  dim3 gk(Dn / 64, (Bn * Tn) / 64);        // 6 x 98 blocks, 128 threads

  for (int l = 0; l < Llay; ++l) {
    const float *cwq = L_in(l,0),  *cbq = L_in(l,1);
    const float *cwk = L_in(l,2),  *cbk = L_in(l,3);
    const float *cwv = L_in(l,4),  *cbv = L_in(l,5);
    const float *cwo = L_in(l,6),  *cbo = L_in(l,7);
    const float *swq = L_in(l,8),  *sbq = L_in(l,9);
    const float *swk = L_in(l,10), *sbk = L_in(l,11);
    const float *swv = L_in(l,12), *sbv = L_in(l,13);
    const float *swo = L_in(l,14), *sbo = L_in(l,15);
    const float *f1w = L_in(l,16), *f1b = L_in(l,17);
    const float *f2w = L_in(l,18), *f2b = L_in(l,19);
    const float *g1  = L_in(l,20), *b1  = L_in(l,21);
    const float *g2  = L_in(l,22), *b2  = L_in(l,23);
    const float *g3  = L_in(l,24), *b3  = L_in(l,25);

    // big K/V projections via LDS-staged WMMA (bf16 in, f32 acc)
    acseg_convT<<<(Dn * Dn + 255) / 256, 256, 0, stream>>>(cwk, wTk, Dn, Dn);
    acseg_convT<<<(Dn * Dn + 255) / 256, 256, 0, stream>>>(cwv, wTv, Dn, Dn);
    acseg_gemm_wmma<<<gk, 128, 0, stream>>>(xb, wTk, cbk, kbuf);
    acseg_gemm_wmma<<<gk, 128, 0, stream>>>(xb, wTv, cbv, vbuf);

    // cross attention on prototypes
    acseg_lin<<<(MN + 255) / 256, 256, 0, stream>>>(outbuf, cwq, cbq, nullptr, nullptr, qbuf, Bn*Kn, Dn, Dn, 0);
    acseg_attn<<<dim3(Kn, Hn, Bn), 256, 0, stream>>>(qbuf, kbuf, vbuf, obuf, Tn);
    acseg_lin<<<(MN + 255) / 256, 256, 0, stream>>>(obuf, cwo, cbo, qbuf, outbuf, t1, Bn*Kn, Dn, Dn, 0);
    acseg_ln<<<Bn * Kn, 128, 0, stream>>>(t1, g1, b1, x1);

    // self attention (5 tokens)
    acseg_lin<<<(MN + 255) / 256, 256, 0, stream>>>(x1, swq, sbq, nullptr, nullptr, q2, Bn*Kn, Dn, Dn, 0);
    acseg_lin<<<(MN + 255) / 256, 256, 0, stream>>>(x1, swk, sbk, nullptr, nullptr, k2, Bn*Kn, Dn, Dn, 0);
    acseg_lin<<<(MN + 255) / 256, 256, 0, stream>>>(x1, swv, sbv, nullptr, nullptr, v2, Bn*Kn, Dn, Dn, 0);
    acseg_attn<<<dim3(Kn, Hn, Bn), 256, 0, stream>>>(q2, k2, v2, o2, Kn);
    acseg_lin<<<(MN + 255) / 256, 256, 0, stream>>>(o2, swo, sbo, q2, x1, t2, Bn*Kn, Dn, Dn, 0);
    acseg_ln<<<Bn * Kn, 128, 0, stream>>>(t2, g2, b2, x2);

    // FFN
    acseg_lin<<<(MN2 + 255) / 256, 256, 0, stream>>>(x2, f1w, f1b, nullptr, nullptr, hb, Bn*Kn, Dn, DFFn, 1);
    acseg_lin<<<(MN + 255) / 256, 256, 0, stream>>>(hb, f2w, f2b, x2, nullptr, t3, Bn*Kn, DFFn, Dn, 0);
    acseg_ln<<<Bn * Kn, 128, 0, stream>>>(t3, g3, b3, outbuf);
  }

  // affinity graph: raw A -> Kdeg/m/kscal -> scaled W (deterministic)
  acseg_affinity<<<dim3((Tn + 63) / 64, (Tn + 63) / 64, Bn), 128, 0, stream>>>(xn, xcn, Wout);
  acseg_rowsum<<<dim3(Tn, Bn), 256, 0, stream>>>(Wout, Kdeg);
  acseg_finalize<<<Bn, 256, 0, stream>>>(Kdeg, Mout, scl);
  size_t tot = (size_t)Bn * Tn * Tn;
  acseg_wscale<<<(unsigned)((tot + 255) / 256), 256, 0, stream>>>(Wout, scl);

  // pixel assignment
  acseg_normrows<<<Bn * Kn, 128, 0, stream>>>(outbuf, concn);
  acseg_sassign<<<dim3(Tn, Bn), 128, 0, stream>>>(x, concn, S, Aout);
  acseg_delta<<<dim3(Tn, Bn), 256, 0, stream>>>(S, Dout);
}
